// HungarianMatcher_12876311953748
// MI455X (gfx1250) — matrix operand order repaired and checked by reading, not compiled, not run on hardware
//
#include <hip/hip_runtime.h>
#include <hip/hip_bf16.h>

typedef __attribute__((ext_vector_type(16))) _Float16 v16h;
typedef __attribute__((ext_vector_type(8)))  float    v8f;

namespace {
constexpr int NQ = 300;   // queries per batch
constexpr int NC = 13;    // classes
constexpr int NTILES = (NQ + 15) / 16;   // 19 tiles of 16 queries
constexpr float W_CLASS = 1.0f;
constexpr float W_BBOX  = 5.0f;
constexpr float W_GIOU  = 2.0f;
}

struct Tgt {
    float cx, cy, w, h;       // original cxcywh (for L1)
    float x0, y0, x1, y1;     // xyxy
    float area;
};

__device__ __forceinline__ Tgt make_tgt(const float4 tb) {
    Tgt t;
    t.cx = tb.x; t.cy = tb.y; t.w = tb.z; t.h = tb.w;
    t.x0 = tb.x - 0.5f * tb.z;
    t.y0 = tb.y - 0.5f * tb.w;
    t.x1 = tb.x + 0.5f * tb.z;
    t.y1 = tb.y + 0.5f * tb.w;
    t.area = tb.z * tb.w;
    return t;
}

__device__ __forceinline__ float frcp(float x) { return __builtin_amdgcn_rcpf(x); }

// L1 + GIoU terms combined with the WMMA-produced class probability.
__device__ __forceinline__ float pair_cost(const Tgt& t, float prob, const float4& pb,
                                           float px0, float py0, float px1, float py1,
                                           float parea) {
    const float l1 = fabsf(pb.x - t.cx) + fabsf(pb.y - t.cy) +
                     fabsf(pb.z - t.w)  + fabsf(pb.w - t.h);
    const float ix0 = fmaxf(px0, t.x0);
    const float iy0 = fmaxf(py0, t.y0);
    const float ix1 = fminf(px1, t.x1);
    const float iy1 = fminf(py1, t.y1);
    const float iw  = fmaxf(ix1 - ix0, 0.0f);
    const float ih  = fmaxf(iy1 - iy0, 0.0f);
    const float inter = iw * ih;
    const float uni   = parea + t.area - inter;
    const float iou   = inter * frcp(uni);
    const float ex0 = fminf(px0, t.x0);
    const float ey0 = fminf(py0, t.y0);
    const float ex1 = fmaxf(px1, t.x1);
    const float ey1 = fmaxf(py1, t.y1);
    const float earea = (ex1 - ex0) * (ey1 - ey0);   // enclosing box is non-degenerate
    const float giou  = iou - (earea - uni) * frcp(earea);
    return -W_CLASS * prob + W_BBOX * l1 - W_GIOU * giou;
}

__launch_bounds__(256, 2)
__global__ void matcher_cost_kernel(const float* __restrict__ pred_logits,
                                    const float* __restrict__ pred_boxes,
                                    const float* __restrict__ tgt_boxes,
                                    const int*   __restrict__ tgt_ids,
                                    float* __restrict__ out)
{
    const int b    = blockIdx.x;
    const int tid  = threadIdx.x;
    const int wave = tid >> 5;
    const int lane = tid & 31;

    const int tlow  = lane & 15;   // D-matrix column N = target index (mod 16)
    const int khalf = lane >> 4;   // which K-half of A/B this lane carries
    const int m     = tlow;        // A-matrix row M for this lane

    // Reference slices [:, :, :T] => every batch matched against batch 0's targets.
    // Each lane only ever touches targets tlow and tlow+16 -> hold them in registers.
    const Tgt t0 = make_tgt(reinterpret_cast<const float4*>(tgt_boxes)[tlow]);
    const Tgt t1 = make_tgt(reinterpret_cast<const float4*>(tgt_boxes)[tlow + 16]);
    const int id0 = tgt_ids[tlow];
    const int id1 = tgt_ids[tlow + 16];

    // One-hot B fragments (K=32 x N=16, f16), reused across all q tiles.
    // Lanes 0-15 carry K=0..15 for column N=lane; lanes 16-31 carry K=16..31 (all zero: classes < 16).
    v16h bm0, bm1;
#pragma unroll
    for (int j = 0; j < 16; ++j) {
        bm0[j] = (khalf == 0 && id0 == j) ? (_Float16)1.0f : (_Float16)0.0f;
        bm1[j] = (khalf == 0 && id1 == j) ? (_Float16)1.0f : (_Float16)0.0f;
    }

    for (int tile = wave; tile < NTILES; tile += 8) {
        if (tile + 8 < NTILES) {   // wave-uniform; pull next tile's logits toward the caches
            __builtin_prefetch(pred_logits + (size_t)(b * NQ + (tile + 8) * 16 + m) * NC, 0, 1);
        }

        const int q  = tile * 16 + m;
        const int qc = q < NQ ? q : NQ - 1;           // clamp tail reads (tail rows never stored)
        const float* lg = pred_logits + (size_t)(b * NQ + qc) * NC;

        // 13-class softmax for row M = m of this tile
        float x[NC];
#pragma unroll
        for (int c = 0; c < NC; ++c) x[c] = lg[c];
        float mx = x[0];
#pragma unroll
        for (int c = 1; c < NC; ++c) mx = fmaxf(mx, x[c]);
        float sum = 0.0f;
#pragma unroll
        for (int c = 0; c < NC; ++c) { x[c] = __expf(x[c] - mx); sum += x[c]; }
        const float inv = frcp(sum);

        // A fragment: 16x32 f16. Lanes 0-15: elems 0..15 = K 0..15 (zero past class 12).
        // Lanes 16-31: elems 0..15 = K 16..31 (zero).
        v16h am;
#pragma unroll
        for (int j = 0; j < 8; ++j) {
            const float plo = x[j] * inv;
            const float phi = (j + 8 < NC) ? x[j + 8] * inv : 0.0f;
            am[j]     = (_Float16)(khalf == 0 ? plo : 0.0f);
            am[j + 8] = (_Float16)(khalf == 0 ? phi : 0.0f);
        }

        // Class-cost GEMM on the matrix pipe: D[q_row, t] = P[q_row, id[t]]
        v8f acc0 = {}, acc1 = {};
        acc0 = __builtin_amdgcn_wmma_f32_16x16x32_f16(false, am, false, bm0,
                                                      (short)0, acc0, false, false);
        acc1 = __builtin_amdgcn_wmma_f32_16x16x32_f16(false, am, false, bm1,
                                                      (short)0, acc1, false, false);

        // D layout: lane holds column N=tlow, VGPR r holds row M = r + 8*khalf.
        auto do_rows = [&](bool guarded) {
#pragma unroll
            for (int r = 0; r < 8; ++r) {
                const int qr  = tile * 16 + r + 8 * khalf;
                const int qrc = qr < NQ ? qr : NQ - 1;
                const float4 pb = reinterpret_cast<const float4*>(pred_boxes)[b * NQ + qrc];
                const float px0 = pb.x - 0.5f * pb.z;
                const float py0 = pb.y - 0.5f * pb.w;
                const float px1 = pb.x + 0.5f * pb.z;
                const float py1 = pb.y + 0.5f * pb.w;
                const float parea = pb.z * pb.w;

                const float c0 = pair_cost(t0, acc0[r], pb, px0, py0, px1, py1, parea);
                const float c1 = pair_cost(t1, acc1[r], pb, px0, py0, px1, py1, parea);

                if (!guarded || qr < NQ) {
                    float* o = out + (((size_t)b * NQ + qr) << 5);  // *32 targets
                    o[tlow]      = c0;   // lanes 0..15 and 16..31 hit distinct q rows
                    o[tlow + 16] = c1;
                }
            }
        };
        if (tile != NTILES - 1) do_rows(false);   // tiles 0..17: branch-free stores
        else                    do_rows(true);    // tile 18: guard q >= 300
    }
}

extern "C" void kernel_launch(void* const* d_in, const int* in_sizes, int n_in,
                              void* d_out, int out_size, void* d_ws, size_t ws_size,
                              hipStream_t stream) {
    const float* pred_logits = (const float*)d_in[0];
    const float* pred_boxes  = (const float*)d_in[1];
    const float* tgt_boxes   = (const float*)d_in[2];
    const int*   tgt_ids     = (const int*)d_in[3];
    float* out = (float*)d_out;

    matcher_cost_kernel<<<dim3(64), dim3(256), 0, stream>>>(
        pred_logits, pred_boxes, tgt_boxes, tgt_ids, out);
}
